// EnhancedCoordinate3DPredictor_16303695855964
// MI455X (gfx1250) — matrix operand order repaired and checked by reading, not compiled, not run on hardware
//
#include <hip/hip_runtime.h>
#include <hip/hip_bf16.h>

typedef _Float16 half_t;
typedef __attribute__((ext_vector_type(16))) half_t v16h;
typedef __attribute__((ext_vector_type(8)))  half_t v8h;
typedef __attribute__((ext_vector_type(8)))  float  v8f;
typedef __attribute__((ext_vector_type(4)))  int    v4i;

#define NN_    4096
#define NE_    16384
#define HID_   128
#define HEADS_ 8
#define LAY_   4
#define EDIM_  32
#define HH_    1024   /* HEADS_*HID_ */
#define NB_    8
#define NPG_   512

// ---------------------------------------------------------------------------
// Async global->LDS copy of 16 bytes (one b128 per lane), with safe fallback.
#if defined(__has_builtin)
#if __has_builtin(__builtin_amdgcn_global_load_async_to_lds_b128)
#define HAVE_ASYNC_LDS 1
#endif
#endif

__device__ __forceinline__ void cpy16_g2l(const half_t* gp, half_t* lp) {
#ifdef HAVE_ASYNC_LDS
    typedef __attribute__((address_space(1))) v4i gv4i;
    typedef __attribute__((address_space(3))) v4i lv4i;
    __builtin_amdgcn_global_load_async_to_lds_b128(
        (gv4i*)(void*)gp, (lv4i*)(void*)lp, 0, 0);
#else
    *(v8h*)lp = *(const v8h*)gp;   // global_load_b128 + ds_store_b128
#endif
}

__device__ __forceinline__ void async_wait_all() {
#ifdef HAVE_ASYNC_LDS
#if __has_builtin(__builtin_amdgcn_s_wait_asynccnt)
    __builtin_amdgcn_s_wait_asynccnt(0);
#else
    asm volatile("s_wait_asynccnt 0x0" ::: "memory");
#endif
#endif
}

// ---------------------------------------------------------------------------
// f32 -> f16 conversion
__global__ void k_cvt(const float* __restrict__ s, half_t* __restrict__ d, long n) {
    long i = (long)blockIdx.x * blockDim.x + threadIdx.x;
    if (i < n) d[i] = (half_t)s[i];
}

// gelu(x) then f32 -> f16
__global__ void k_gelu16(const float* __restrict__ s, half_t* __restrict__ d, long n) {
    long i = (long)blockIdx.x * blockDim.x + threadIdx.x;
    if (i < n) {
        float x = s[i];
        d[i] = (half_t)(0.5f * x * (1.f + erff(x * 0.70710678118654752f)));
    }
}

// h = atom_emb[x]; also f16 mirror
__global__ void k_embed(const int* __restrict__ x, const float* __restrict__ emb,
                        float* __restrict__ h, half_t* __restrict__ h16) {
    int i = blockIdx.x * blockDim.x + threadIdx.x;
    if (i >= NN_ * HID_) return;
    int n = i / HID_, d = i % HID_;
    float v = emb[x[n] * HID_ + d];
    h[i] = v; h16[i] = (half_t)v;
}

// e = edge_attr @ edge_w + edge_b  (E x 32)
__global__ void k_edge_embed(const float* __restrict__ ea, const float* __restrict__ ew,
                             const float* __restrict__ eb,
                             float* __restrict__ e, half_t* __restrict__ e16) {
    int i = blockIdx.x * blockDim.x + threadIdx.x;
    if (i >= NE_ * EDIM_) return;
    int ed = i / EDIM_, j = i % EDIM_;
    float v = ea[ed] * ew[j] + eb[j];
    e[i] = v; e16[i] = (half_t)v;
}

// ---------------------------------------------------------------------------
// WMMA GEMM: C[M,N] (f32) = A[M,K] (f16 row-major) @ B[K,N] (f16 row-major) + bias
// Block = PW/16 waves; block computes a 16 x PW tile of C.
// A tile staged via async copies; B panel staged TRANSPOSED (BT[col][k]) so the
// hot loop is 4x ds_load_b128 + v_wmma per K-step (no element packing).
// Requires M%16==0, N%PW==0, PW%16==0, K%32==0.
__global__ void k_gemm(const half_t* __restrict__ A, const half_t* __restrict__ B,
                       const float* __restrict__ bias, float* __restrict__ C,
                       int M, int N, int K, int PW,
                       long long aStride, long long bStride, long long cStride) {
    extern __shared__ half_t lds[];
    A += (long long)blockIdx.z * aStride;
    B += (long long)blockIdx.z * bStride;
    C += (long long)blockIdx.z * cStride;
    const int tid  = threadIdx.x;
    const int lane = tid & 31;
    const int wave = tid >> 5;
    const int lo = lane & 15;     // row (A/C) / column (B/C)
    const int hi = lane >> 4;     // half-wave selector
    const int mt = blockIdx.x;
    const int cp = blockIdx.y * PW;
    const int KPS = K + 8;        // padded K-stride (halves): rows stay 16B-aligned
    half_t* ldsA  = lds;                // 16 x KPS  (A tile, row-major)
    half_t* ldsBT = lds + 16 * KPS;     // PW x KPS  (B^T panel: [col][k])

    // Stage A tile: 16 rows x K halves, async 16B chunks.
    const int aChunks = 16 * (K >> 3);
    for (int c = tid; c < aChunks; c += blockDim.x) {
        int r  = c / (K >> 3);
        int cc = (c % (K >> 3)) << 3;
        cpy16_g2l(A + (long long)(mt * 16 + r) * K + cc, ldsA + r * KPS + cc);
    }
    // Stage B panel transposed: read 8 consecutive columns of row k (one b128),
    // scatter into 8 BT rows at offset k.
    const int bChunks = K * (PW >> 3);
    for (int c = tid; c < bChunks; c += blockDim.x) {
        int k  = c / (PW >> 3);
        int cc = (c % (PW >> 3)) << 3;
        v8h t = *(const v8h*)(B + (long long)k * N + cp + cc);
#pragma unroll
        for (int j = 0; j < 8; ++j) ldsBT[(cc + j) * KPS + k] = t[j];
    }
    async_wait_all();
    __syncthreads();

    v8f acc = {};
    const half_t* arow  = ldsA  + lo * KPS;
    const half_t* btrow = ldsBT + (wave * 16 + lo) * KPS;
    for (int kk = 0; kk < K; kk += 32) {
        // A fragment: row M=lo; j<8 -> K=kk+hi*8+j ; j>=8 -> K=kk+16+hi*8+(j-8)
        union { v16h v; v8h h2[2]; } a, b;
        a.h2[0] = *(const v8h*)(arow + kk + hi * 8);
        a.h2[1] = *(const v8h*)(arow + kk + 16 + hi * 8);
        // B fragment: col N=lo; j<8 -> K=kk+hi*8+j ; j>=8 -> K=kk+16+hi*8+(j-8) (from BT[col][K])
        b.h2[0] = *(const v8h*)(btrow + kk + hi * 8);
        b.h2[1] = *(const v8h*)(btrow + kk + 16 + hi * 8);
        acc = __builtin_amdgcn_wmma_f32_16x16x32_f16(false, a.v, false, b.v,
                                                     (short)0, acc, false, false);
    }
    int col = cp + wave * 16 + lo;
    float bvadd = bias ? bias[col] : 0.f;
#pragma unroll
    for (int r = 0; r < 8; ++r)
        C[(long long)(mt * 16 + r + hi * 8) * N + col] = acc[r] + bvadd;
}

// ---------------------------------------------------------------------------
// per-layer scratch init: m=-inf, z=0, agg=0
__global__ void k_init_layer(float* __restrict__ m, float* __restrict__ z,
                             float* __restrict__ agg) {
    long i = (long)blockIdx.x * blockDim.x + threadIdx.x;
    if (i < NN_ * HEADS_) { m[i] = -1e30f; z[i] = 0.f; }
    if (i < (long)NN_ * HH_) agg[i] = 0.f;
}

// logits[e,h] = dot(q[dst,h,:], k[src,h,:]+ee[e,h,:]) / sqrt(HID)
__global__ void k_edge_logits(const int* __restrict__ ei, const float* __restrict__ q,
                              const float* __restrict__ k, const float* __restrict__ ee,
                              float* __restrict__ logits) {
    int e = blockIdx.x;
    int h = threadIdx.x >> 5;
    int l = threadIdx.x & 31;
    int src = ei[e], dst = ei[NE_ + e];
    const float* qp = q + (long long)dst * HH_ + h * HID_;
    const float* kp = k + (long long)src * HH_ + h * HID_;
    const float* ep = ee + (long long)e * HH_ + h * HID_;
    float s = 0.f;
    for (int d = l; d < HID_; d += 32) s += qp[d] * (kp[d] + ep[d]);
#pragma unroll
    for (int mk = 16; mk > 0; mk >>= 1) s += __shfl_xor(s, mk, 32);
    if (l == 0) logits[e * HEADS_ + h] = s * 0.08838834764831845f; // 1/sqrt(128)
}

__device__ __forceinline__ void atomicMaxF(float* addr, float val) {
    unsigned* ua = (unsigned*)addr;
    unsigned old = *ua;
    while (true) {
        float f = __uint_as_float(old);
        if (f >= val) break;
        unsigned assumed = old;
        old = atomicCAS(ua, assumed, __float_as_uint(val));
        if (old == assumed) break;
    }
}

__global__ void k_segmax(const int* __restrict__ ei, const float* __restrict__ logits,
                         float* __restrict__ m) {
    int i = blockIdx.x * blockDim.x + threadIdx.x;
    if (i >= NE_ * HEADS_) return;
    int e = i / HEADS_, h = i % HEADS_;
    int dst = ei[NE_ + e];
    atomicMaxF(&m[dst * HEADS_ + h], logits[i]);
}

__global__ void k_edge_exp(const int* __restrict__ ei, float* __restrict__ logits,
                           const float* __restrict__ m, float* __restrict__ z) {
    int i = blockIdx.x * blockDim.x + threadIdx.x;
    if (i >= NE_ * HEADS_) return;
    int e = i / HEADS_, h = i % HEADS_;
    int dst = ei[NE_ + e];
    float a = expf(logits[i] - m[dst * HEADS_ + h]);
    logits[i] = a;
    atomicAdd(&z[dst * HEADS_ + h], a);
}

// agg[dst,h,d] += (a/z) * (v[src,h,d] + ee[e,h,d])
__global__ void k_edge_agg(const int* __restrict__ ei, const float* __restrict__ aexp,
                           const float* __restrict__ z, const float* __restrict__ v,
                           const float* __restrict__ ee, float* __restrict__ agg) {
    long i = (long)blockIdx.x * blockDim.x + threadIdx.x;
    if (i >= (long)NE_ * HH_) return;
    int e = (int)(i / HH_);
    int rem = (int)(i % HH_);
    int h = rem / HID_;
    int src = ei[e], dst = ei[NE_ + e];
    float w = aexp[e * HEADS_ + h] / (z[dst * HEADS_ + h] + 1e-16f);
    float vj = v[(long long)src * HH_ + rem] + ee[i];
    atomicAdd(&agg[(long long)dst * HH_ + rem], w * vj);
}

// out = mean_h(agg) + skip; h = gelu(LN(out + res)); LDS tree reductions
__global__ void k_post(const float* __restrict__ agg, const float* __restrict__ skip,
                       const float* __restrict__ lng, const float* __restrict__ lnb,
                       float* __restrict__ h, half_t* __restrict__ h16) {
    int n = blockIdx.x, d = threadIdx.x;
    float s = 0.f;
    for (int hh = 0; hh < HEADS_; ++hh) s += agg[(long long)n * HH_ + hh * HID_ + d];
    float val = s * 0.125f + skip[n * HID_ + d] + h[n * HID_ + d];
    __shared__ float red[HID_];
    red[d] = val; __syncthreads();
    for (int st = 64; st > 0; st >>= 1) { if (d < st) red[d] += red[d + st]; __syncthreads(); }
    float mean = red[0] * (1.f / HID_); __syncthreads();
    float c = val - mean;
    red[d] = c * c; __syncthreads();
    for (int st = 64; st > 0; st >>= 1) { if (d < st) red[d] += red[d + st]; __syncthreads(); }
    float var = red[0] * (1.f / HID_);
    float xln = c * rsqrtf(var + 1e-5f) * lng[d] + lnb[d];
    float g = 0.5f * xln * (1.f + erff(xln * 0.70710678118654752f));
    h[n * HID_ + d] = g; h16[n * HID_ + d] = (half_t)g;
}

// ---------------------------------------------------------------------------
// MHA: split qkv into q padded (K=16 -> 32 zero pad), k^T padded, v; all f16
__global__ void k_mha_prep(const float* __restrict__ qkv, half_t* __restrict__ qpad,
                           half_t* __restrict__ kT, half_t* __restrict__ vv) {
    long i = (long)blockIdx.x * blockDim.x + threadIdx.x;
    if (i >= (long)NN_ * HEADS_ * 32) return;
    int dp = (int)(i & 31);
    int h  = (int)((i >> 5) & 7);
    int n  = (int)(i >> 8);
    int b = n / NPG_, p = n % NPG_;
    int bh = b * HEADS_ + h;
    float qv = 0.f, kv = 0.f;
    if (dp < 16) {
        qv = qkv[(long long)n * 384 + h * 16 + dp];
        kv = qkv[(long long)n * 384 + 128 + h * 16 + dp];
        vv[((long long)bh * NPG_ + p) * 16 + dp] =
            (half_t)qkv[(long long)n * 384 + 256 + h * 16 + dp];
    }
    qpad[((long long)bh * NPG_ + p) * 32 + dp] = (half_t)qv;
    kT[(long long)bh * 32 * NPG_ + (long long)dp * NPG_ + p] = (half_t)kv;
}

// row softmax over 512 with scale 1/sqrt(16)=0.25; writes f16 probs
__global__ void k_softmax(const float* __restrict__ att, half_t* __restrict__ att16) {
    int row = blockIdx.x;
    int t = threadIdx.x;  // 256
    const float* a = att + (long long)row * NPG_;
    float v0 = a[t] * 0.25f, v1 = a[t + 256] * 0.25f;
    __shared__ float red[256];
    red[t] = fmaxf(v0, v1); __syncthreads();
    for (int s = 128; s > 0; s >>= 1) { if (t < s) red[t] = fmaxf(red[t], red[t + s]); __syncthreads(); }
    float mx = red[0]; __syncthreads();
    float e0 = expf(v0 - mx), e1 = expf(v1 - mx);
    red[t] = e0 + e1; __syncthreads();
    for (int s = 128; s > 0; s >>= 1) { if (t < s) red[t] += red[t + s]; __syncthreads(); }
    float inv = 1.f / red[0];
    att16[(long long)row * NPG_ + t] = (half_t)(e0 * inv);
    att16[(long long)row * NPG_ + t + 256] = (half_t)(e1 * inv);
}

// (b,h,p,16) -> (n, h*16+d); also f16 mirror
__global__ void k_merge(const float* __restrict__ obuf, float* __restrict__ o,
                        half_t* __restrict__ o16) {
    int i = blockIdx.x * blockDim.x + threadIdx.x;
    if (i >= NN_ * HID_) return;
    int n = i / HID_, r = i % HID_;
    int h = r / 16, d = r % 16;
    int b = n / NPG_, p = n % NPG_;
    float v = obuf[((long long)(b * HEADS_ + h) * NPG_ + p) * 16 + d];
    o[i] = v; o16[i] = (half_t)v;
}

// final head: out[n,3] = gelu_y2 @ w3 + b3 (K=64, N=3)
__global__ void k_head(const half_t* __restrict__ t2g, const float* __restrict__ w3,
                       const float* __restrict__ b3, float* __restrict__ out) {
    int n = blockIdx.x * blockDim.x + threadIdx.x;
    if (n >= NN_) return;
    float a0 = b3[0], a1 = b3[1], a2 = b3[2];
    for (int d = 0; d < 64; ++d) {
        float g = (float)t2g[n * 64 + d];
        a0 += g * w3[d * 3 + 0];
        a1 += g * w3[d * 3 + 1];
        a2 += g * w3[d * 3 + 2];
    }
    out[n * 3 + 0] = a0; out[n * 3 + 1] = a1; out[n * 3 + 2] = a2;
}

// ---------------------------------------------------------------------------
extern "C" void kernel_launch(void* const* d_in, const int* in_sizes, int n_in,
                              void* d_out, int out_size, void* d_ws, size_t ws_size,
                              hipStream_t stream) {
    (void)in_sizes; (void)n_in; (void)out_size; (void)ws_size;
    const int*   x        = (const int*)  d_in[0];
    const int*   ei       = (const int*)  d_in[1];
    const float* edge_at  = (const float*)d_in[2];
    const float* atom_emb = (const float*)d_in[4];
    const float* edge_w   = (const float*)d_in[5];
    const float* edge_b   = (const float*)d_in[6];
    const float* wq = (const float*)d_in[7];   const float* bq = (const float*)d_in[8];
    const float* wk = (const float*)d_in[9];   const float* bk = (const float*)d_in[10];
    const float* wv = (const float*)d_in[11];  const float* bv = (const float*)d_in[12];
    const float* we = (const float*)d_in[13];
    const float* wskip = (const float*)d_in[14]; const float* bskip = (const float*)d_in[15];
    const float* lng = (const float*)d_in[16]; const float* lnb = (const float*)d_in[17];
    const float* mha_in_w  = (const float*)d_in[18]; const float* mha_in_b  = (const float*)d_in[19];
    const float* mha_out_w = (const float*)d_in[20]; const float* mha_out_b = (const float*)d_in[21];
    const float* w1 = (const float*)d_in[22];  const float* b1 = (const float*)d_in[23];
    const float* w2 = (const float*)d_in[24];  const float* b2 = (const float*)d_in[25];
    const float* w3 = (const float*)d_in[26];  const float* b3 = (const float*)d_in[27];
    float* out = (float*)d_out;

    char* ws = (char*)d_ws;
    size_t off = 0;
    auto alloc = [&](size_t bytes) -> char* {
        char* p = ws + off;
        off = (off + bytes + 255) & ~(size_t)255;
        return p;
    };
    float*  h     = (float*) alloc((size_t)NN_ * HID_ * 4);
    half_t* h16   = (half_t*)alloc((size_t)NN_ * HID_ * 2);
    float*  ebuf  = (float*) alloc((size_t)NE_ * EDIM_ * 4);
    half_t* e16   = (half_t*)alloc((size_t)NE_ * EDIM_ * 2);
    float*  q     = (float*) alloc((size_t)NN_ * HH_ * 4);   // reused: att16 (spans q+kbuf)
    float*  kbuf  = (float*) alloc((size_t)NN_ * HH_ * 4);
    float*  vbuf  = (float*) alloc((size_t)NN_ * HH_ * 4);   // reused: MHA small buffers
    float*  eebuf = (float*) alloc((size_t)NE_ * HH_ * 4);   // reused: att f32
    float*  skip  = (float*) alloc((size_t)NN_ * HID_ * 4);
    float*  logit = (float*) alloc((size_t)NE_ * HEADS_ * 4);
    float*  mbuf  = (float*) alloc((size_t)NN_ * HEADS_ * 4);
    float*  zbuf  = (float*) alloc((size_t)NN_ * HEADS_ * 4);
    float*  agg   = (float*) alloc((size_t)NN_ * HH_ * 4);   // reused: qkvbuf
    half_t* wq16  = (half_t*)alloc((size_t)LAY_ * HID_ * HH_ * 2);
    half_t* wk16  = (half_t*)alloc((size_t)LAY_ * HID_ * HH_ * 2);
    half_t* wv16  = (half_t*)alloc((size_t)LAY_ * HID_ * HH_ * 2);
    half_t* we16  = (half_t*)alloc((size_t)LAY_ * EDIM_ * HH_ * 2);
    half_t* wsk16 = (half_t*)alloc((size_t)LAY_ * HID_ * HID_ * 2);
    half_t* min16 = (half_t*)alloc((size_t)HID_ * 384 * 2);
    half_t* mout16= (half_t*)alloc((size_t)HID_ * HID_ * 2);
    half_t* w1_16 = (half_t*)alloc((size_t)HID_ * HID_ * 2);
    half_t* w2_16 = (half_t*)alloc((size_t)HID_ * 64 * 2);

    // Reused regions (layer-phase buffers are dead by MHA time)
    half_t* att16  = (half_t*)q;                            // 32MB over q+kbuf
    float*  attf   = eebuf;                                 // 64MB
    float*  qkvbuf = agg;                                   // 6MB of 16MB
    char* vr = (char*)vbuf;                                 // 16MB region
    half_t* qpad16 = (half_t*)(vr);
    half_t* kT16   = (half_t*)(vr +  2097152);
    half_t* v16n   = (half_t*)(vr +  4194304);
    float*  obuf   = (float*) (vr +  5242880);
    float*  o      = (float*) (vr +  7340032);
    half_t* o16    = (half_t*)(vr +  9437184);
    float*  mo     = (float*) (vr + 10485760);
    half_t* mo16   = (half_t*)(vr + 12582912);
    float*  t1     = o;       // o f32 dead after o16 built
    half_t* t1g16  = (half_t*)(vr + 13631488);
    float*  t2     = obuf;    // obuf dead after merge
    half_t* t2g16  = (half_t*)(vr + 14680064);

    auto smem = [](int K, int PW) -> size_t {
        return (size_t)(16 + PW) * (K + 8) * 2;
    };

#define CVT(src, dst, n) k_cvt<<<(int)(((n) + 255) / 256), 256, 0, stream>>>((src), (dst), (long)(n))
    CVT(wq, wq16, (long)LAY_ * HID_ * HH_);
    CVT(wk, wk16, (long)LAY_ * HID_ * HH_);
    CVT(wv, wv16, (long)LAY_ * HID_ * HH_);
    CVT(we, we16, (long)LAY_ * EDIM_ * HH_);
    CVT(wskip, wsk16, (long)LAY_ * HID_ * HID_);
    CVT(mha_in_w, min16, (long)HID_ * 384);
    CVT(mha_out_w, mout16, (long)HID_ * HID_);
    CVT(w1, w1_16, (long)HID_ * HID_);
    CVT(w2, w2_16, (long)HID_ * 64);

    k_embed<<<(NN_ * HID_ + 255) / 256, 256, 0, stream>>>(x, atom_emb, h, h16);
    k_edge_embed<<<(NE_ * EDIM_ + 255) / 256, 256, 0, stream>>>(edge_at, edge_w, edge_b, ebuf, e16);

    for (int i = 0; i < LAY_; ++i) {
        const half_t* wq_i = wq16 + (size_t)i * HID_ * HH_;
        const half_t* wk_i = wk16 + (size_t)i * HID_ * HH_;
        const half_t* wv_i = wv16 + (size_t)i * HID_ * HH_;
        const half_t* we_i = we16 + (size_t)i * EDIM_ * HH_;
        const half_t* ws_i = wsk16 + (size_t)i * HID_ * HID_;

        dim3 g1(NN_ / 16, HH_ / 128);   // 256 x 8, PW=128
        k_gemm<<<g1, 256, smem(HID_, 128), stream>>>(h16, wq_i, bq + i * HH_, q,
                                                     NN_, HH_, HID_, 128, 0, 0, 0);
        k_gemm<<<g1, 256, smem(HID_, 128), stream>>>(h16, wk_i, bk + i * HH_, kbuf,
                                                     NN_, HH_, HID_, 128, 0, 0, 0);
        k_gemm<<<g1, 256, smem(HID_, 128), stream>>>(h16, wv_i, bv + i * HH_, vbuf,
                                                     NN_, HH_, HID_, 128, 0, 0, 0);
        dim3 g2(NE_ / 16, HH_ / 128);   // 1024 x 8, K=32
        k_gemm<<<g2, 256, smem(EDIM_, 128), stream>>>(e16, we_i, nullptr, eebuf,
                                                      NE_, HH_, EDIM_, 128, 0, 0, 0);
        dim3 g3(NN_ / 16, 1);
        k_gemm<<<g3, 256, smem(HID_, 128), stream>>>(h16, ws_i, bskip + i * HID_, skip,
                                                     NN_, HID_, HID_, 128, 0, 0, 0);

        k_init_layer<<<(int)(((long)NN_ * HH_ + 255) / 256), 256, 0, stream>>>(mbuf, zbuf, agg);
        k_edge_logits<<<NE_, 256, 0, stream>>>(ei, q, kbuf, eebuf, logit);
        k_segmax<<<(NE_ * HEADS_ + 255) / 256, 256, 0, stream>>>(ei, logit, mbuf);
        k_edge_exp<<<(NE_ * HEADS_ + 255) / 256, 256, 0, stream>>>(ei, logit, mbuf, zbuf);
        k_edge_agg<<<(int)(((long)NE_ * HH_ + 255) / 256), 256, 0, stream>>>(ei, logit, zbuf, vbuf, eebuf, agg);
        k_post<<<NN_, HID_, 0, stream>>>(agg, skip, lng + i * HID_, lnb + i * HID_, h, h16);
    }

    // Batched MHA
    k_gemm<<<dim3(NN_ / 16, 3), 256, smem(HID_, 128), stream>>>(
        h16, min16, mha_in_b, qkvbuf, NN_, 384, HID_, 128, 0, 0, 0);
    k_mha_prep<<<(int)(((long)NN_ * HEADS_ * 32 + 255) / 256), 256, 0, stream>>>(qkvbuf, qpad16, kT16, v16n);
    k_gemm<<<dim3(NPG_ / 16, NPG_ / 128, NB_ * HEADS_), 256, smem(32, 128), stream>>>(
        qpad16, kT16, nullptr, attf, NPG_, NPG_, 32, 128,
        (long long)NPG_ * 32, (long long)32 * NPG_, (long long)NPG_ * NPG_);
    k_softmax<<<NB_ * HEADS_ * NPG_, 256, 0, stream>>>(attf, att16);
    k_gemm<<<dim3(NPG_ / 16, 1, NB_ * HEADS_), 32, smem(NPG_, 16), stream>>>(
        att16, v16n, nullptr, obuf, NPG_, 16, NPG_, 16,
        (long long)NPG_ * NPG_, (long long)NPG_ * 16, (long long)NPG_ * 16);
    k_merge<<<(NN_ * HID_ + 255) / 256, 256, 0, stream>>>(obuf, o, o16);
    k_gemm<<<dim3(NN_ / 16, 1), 256, smem(HID_, 128), stream>>>(
        o16, mout16, mha_out_b, mo, NN_, HID_, HID_, 128, 0, 0, 0);
    CVT(mo, mo16, (long)NN_ * HID_);
    k_gemm<<<dim3(NN_ / 16, 1), 256, smem(HID_, 128), stream>>>(
        mo16, w1_16, b1, t1, NN_, HID_, HID_, 128, 0, 0, 0);
    k_gelu16<<<(NN_ * HID_ + 255) / 256, 256, 0, stream>>>(t1, t1g16, (long)NN_ * HID_);
    k_gemm<<<dim3(NN_ / 16, 1), 128, smem(HID_, 64), stream>>>(
        t1g16, w2_16, b2, t2, NN_, 64, HID_, 64, 0, 0, 0);
    k_gelu16<<<(NN_ * 64 + 255) / 256, 256, 0, stream>>>(t2, t2g16, (long)NN_ * 64);
    k_head<<<(NN_ + 63) / 64, 64, 0, stream>>>(t2g16, w3, b3, out);
#undef CVT
}
